// RNN_13374528160354
// MI455X (gfx1250) — matrix operand order repaired and checked by reading, not compiled, not run on hardware
//
#include <hip/hip_runtime.h>
#include <hip/hip_bf16.h>
#include <math.h>

// ---------------------------------------------------------------------------
// 2-layer LSTM (S=256,B=128,I=512,H=1024) + linear head on h2[4].
// Only timesteps 0..4 are live (output = f(h2[4])), so we compute exactly 5
// steps per layer. All GEMMs run on v_wmma_f32_16x16x32_f16 (f16 in, f32 acc).
// ---------------------------------------------------------------------------

#define S_LIVE 5
#define BB     128
#define II     512
#define HH     1024
#define G4     4096   // 4*H
#define OO     32

typedef __attribute__((ext_vector_type(16))) _Float16 v16h;
typedef __attribute__((ext_vector_type(8)))  _Float16 v8h;
typedef __attribute__((ext_vector_type(8)))  float    v8f;

// A-fragment: 16x32 (MxK) f16 from row-major [rows, ld].
// Lane l in 0..15 -> M = m0 + l, K-subgroup 0; lanes 16..31 -> same M, K-subgroup 8.
// slots 0..7  : K = k0 + kh + s
// slots 8..15 : K = k0 + 16 + kh + s
__device__ __forceinline__ v16h load_frag_a(const _Float16* __restrict__ base,
                                            int ld, int m0, int k0, int lane) {
    int m  = m0 + (lane & 15);
    int kh = (lane >> 4) << 3;              // 0 or 8
    const _Float16* p = base + (size_t)m * ld + k0 + kh;
    v8h lo = *(const v8h*)(p);
    v8h hi = *(const v8h*)(p + 16);
    v16h a;
#pragma unroll
    for (int s = 0; s < 8; ++s) { a[s] = lo[s]; a[s + 8] = hi[s]; }
    return a;
}

// B-fragment: 32x16 (KxN) f16 sourced from row-major W[4H, ld]; column N of B is
// row (n0+N) of W, so h @ W^T needs no transpose. Lane l: N = n0 + (l&15),
// slots 0..7 -> K = k0 + 8*(l>>4) + s; slots 8..15 -> K = k0 + 16 + 8*(l>>4) + (s-8).
__device__ __forceinline__ v16h load_frag_b(const _Float16* __restrict__ W,
                                            int ld, int n0, int k0, int lane) {
    int n  = n0 + (lane & 15);
    int kk = k0 + ((lane >> 4) << 3);       // +0 or +8
    const _Float16* p = W + (size_t)n * ld + kk;
    v8h lo = *(const v8h*)(p);
    v8h hi = *(const v8h*)(p + 16);
    v16h b;
#pragma unroll
    for (int s = 0; s < 8; ++s) { b[s] = lo[s]; b[s + 8] = hi[s]; }
    return b;
}

__device__ __forceinline__ v8f wmma_f16(v16h a, v16h b, v8f c) {
    return __builtin_amdgcn_wmma_f32_16x16x32_f16(false, a, false, b,
                                                  (short)0, c, false, false);
}

// ---------------------------------------------------------------------------
__global__ void k_cvt_f16(const float* __restrict__ src, _Float16* __restrict__ dst, int n) {
    int i = blockIdx.x * blockDim.x + threadIdx.x;
    int stride = gridDim.x * blockDim.x;
    for (; i < n; i += stride) dst[i] = (_Float16)src[i];
}

__global__ void k_zero_u32(unsigned int* __restrict__ p, int n) {
    int i = blockIdx.x * blockDim.x + threadIdx.x;
    int stride = gridDim.x * blockDim.x;
    for (; i < n; i += stride) p[i] = 0u;
}

// ---------------------------------------------------------------------------
// XG = X @ W^T + (b_ih + b_hh).  X: [640, Kdim] f16 row-major,
// W: [4096, Kdim] f16 row-major, XG: [640, 4096] f32.
// One wave computes a 16(M) x 64(N) strip (A-fragment reused 4x).
// Launch: 2560 waves = 40 m-tiles * 64 n-groups; 128 threads/block, 640 blocks.
__global__ void k_xg_gemm(const _Float16* __restrict__ X, const _Float16* __restrict__ W,
                          const float* __restrict__ bih, const float* __restrict__ bhh,
                          float* __restrict__ XG, int Kdim) {
    int wave = (blockIdx.x * blockDim.x + threadIdx.x) >> 5;
    int lane = threadIdx.x & 31;
    int mt = wave >> 6;              // 0..39
    int ng = wave & 63;              // 0..63
    int m0 = mt << 4;
    int n0 = ng << 6;

    v8f acc[4] = {};
    for (int k0 = 0; k0 < Kdim; k0 += 32) {
        v16h a = load_frag_a(X, Kdim, m0, k0, lane);
#pragma unroll
        for (int t = 0; t < 4; ++t)
            acc[t] = wmma_f16(a, load_frag_b(W, Kdim, n0 + (t << 4), k0, lane), acc[t]);
    }

    int nl = lane & 15;
    int mb = m0 + ((lane >> 4) << 3);
#pragma unroll
    for (int t = 0; t < 4; ++t) {
        int n = n0 + (t << 4) + nl;
        float bias = bih[n] + bhh[n];
#pragma unroll
        for (int j = 0; j < 8; ++j)
            XG[(size_t)(mb + j) * G4 + n] = acc[t][j] + bias;
    }
}

// ---------------------------------------------------------------------------
// One LSTM timestep: G = xg_t + h_prev @ Whh^T, then fused cell update.
// One wave owns a 16x16 (m,n) tile of H and computes it for ALL FOUR gates
// (same A-fragment; B columns at n, n+1024, n+2048, n+3072), then applies the
// nonlinearity directly -- the 4096-wide gate tensor never exists in memory.
// Launch: 512 waves = 8 m-tiles * 64 n-tiles; 128 threads/block, 128 blocks.
__global__ void k_lstm_step(const float* __restrict__ xg, const _Float16* __restrict__ hprev,
                            const _Float16* __restrict__ whh, float* __restrict__ c,
                            _Float16* __restrict__ hout, _Float16* __restrict__ hcopy,
                            float* __restrict__ hf32) {
    int wave = (blockIdx.x * blockDim.x + threadIdx.x) >> 5;
    int lane = threadIdx.x & 31;
    int mt = wave >> 6;              // 0..7
    int nt = wave & 63;              // 0..63
    int m0 = mt << 4, n0 = nt << 4;

    v8f acc[4] = {};
    for (int k0 = 0; k0 < HH; k0 += 32) {
        v16h a = load_frag_a(hprev, HH, m0, k0, lane);
#pragma unroll
        for (int g = 0; g < 4; ++g)
            acc[g] = wmma_f16(a, load_frag_b(whh, HH, (g << 10) + n0, k0, lane), acc[g]);
    }

    int n  = n0 + (lane & 15);
    int mb = m0 + ((lane >> 4) << 3);
#pragma unroll
    for (int j = 0; j < 8; ++j) {
        int m = mb + j;
        const float* xr = xg + (size_t)m * G4;
        float gi = acc[0][j] + xr[n];
        float gf = acc[1][j] + xr[HH + n];
        float gc = acc[2][j] + xr[2 * HH + n];
        float go = acc[3][j] + xr[3 * HH + n];
        float si = 1.0f / (1.0f + expf(-gi));
        float sf = 1.0f / (1.0f + expf(-gf));
        float so = 1.0f / (1.0f + expf(-go));
        size_t idx = (size_t)m * HH + n;
        float cn = sf * c[idx] + si * tanhf(gc);
        c[idx] = cn;
        float hn = so * tanhf(cn);
        hout[idx] = (_Float16)hn;
        if (hcopy) hcopy[idx] = (_Float16)hn;
        if (hf32)  hf32[idx]  = hn;
    }
}

// ---------------------------------------------------------------------------
// out[b,o] = sum_h h2[b,h] * w_out[o,h] + b_out[o]   (128x32, K=1024, exact fp32)
__global__ void k_head(const float* __restrict__ h, const float* __restrict__ wout,
                       const float* __restrict__ bout, float* __restrict__ out) {
    int idx = blockIdx.x * blockDim.x + threadIdx.x;   // 0..4095
    int b = idx >> 5, o = idx & 31;
    const float4* hp = (const float4*)(h + (size_t)b * HH);
    const float4* wp = (const float4*)(wout + (size_t)o * HH);
    float s = 0.0f;
#pragma unroll 4
    for (int i = 0; i < HH / 4; ++i) {
        float4 x = hp[i], w = wp[i];
        s = fmaf(x.x, w.x, fmaf(x.y, w.y, fmaf(x.z, w.z, fmaf(x.w, w.w, s))));
    }
    out[(size_t)b * OO + o] = s + bout[o];
}

// ---------------------------------------------------------------------------
extern "C" void kernel_launch(void* const* d_in, const int* in_sizes, int n_in,
                              void* d_out, int out_size, void* d_ws, size_t ws_size,
                              hipStream_t stream) {
    const float* x    = (const float*)d_in[0];
    const float* wih0 = (const float*)d_in[1];
    const float* whh0 = (const float*)d_in[2];
    const float* bih0 = (const float*)d_in[3];
    const float* bhh0 = (const float*)d_in[4];
    const float* wih1 = (const float*)d_in[5];
    const float* whh1 = (const float*)d_in[6];
    const float* bih1 = (const float*)d_in[7];
    const float* bhh1 = (const float*)d_in[8];
    const float* wout = (const float*)d_in[9];
    const float* bout = (const float*)d_in[10];
    float* out = (float*)d_out;

    // workspace carve-out (256B aligned)
    char* ws = (char*)d_ws;
    size_t off = 0;
    auto take = [&](size_t bytes) {
        off = (off + 255) & ~(size_t)255;
        char* p = ws + off;
        off += bytes;
        return p;
    };
    _Float16* wih0h = (_Float16*)take((size_t)G4 * II * 2);
    _Float16* whh0h = (_Float16*)take((size_t)G4 * HH * 2);
    _Float16* wih1h = (_Float16*)take((size_t)G4 * HH * 2);
    _Float16* whh1h = (_Float16*)take((size_t)G4 * HH * 2);
    _Float16* xh    = (_Float16*)take((size_t)S_LIVE * BB * II * 2);
    _Float16* h1all = (_Float16*)take((size_t)S_LIVE * BB * HH * 2);
    _Float16* hA    = (_Float16*)take((size_t)BB * HH * 2);
    _Float16* hB    = (_Float16*)take((size_t)BB * HH * 2);
    float*    cbuf  = (float*)take((size_t)BB * HH * 4);
    float*    xgbuf = (float*)take((size_t)S_LIVE * BB * G4 * 4);   // reused by both layers
    float*    h2f32 = (float*)take((size_t)BB * HH * 4);

    // ---- convert weights + live input slice to f16 (28MB f16 -> L2 resident)
    k_cvt_f16<<<2048, 256, 0, stream>>>(wih0, wih0h, G4 * II);
    k_cvt_f16<<<2048, 256, 0, stream>>>(whh0, whh0h, G4 * HH);
    k_cvt_f16<<<2048, 256, 0, stream>>>(wih1, wih1h, G4 * HH);
    k_cvt_f16<<<2048, 256, 0, stream>>>(whh1, whh1h, G4 * HH);
    k_cvt_f16<<<640,  256, 0, stream>>>(x, xh, S_LIVE * BB * II);   // x[0:5] is contiguous

    // ---- layer 0
    k_zero_u32<<<256, 256, 0, stream>>>((unsigned int*)hA,   BB * HH / 2);
    k_zero_u32<<<256, 256, 0, stream>>>((unsigned int*)cbuf, BB * HH);
    k_xg_gemm<<<640, 128, 0, stream>>>(xh, wih0h, bih0, bhh0, xgbuf, II);
    for (int t = 0; t < S_LIVE; ++t) {
        const _Float16* hp = (t & 1) ? hB : hA;
        _Float16*       ho = (t & 1) ? hA : hB;
        k_lstm_step<<<128, 128, 0, stream>>>(xgbuf + (size_t)t * BB * G4, hp, whh0h,
                                             cbuf, ho, h1all + (size_t)t * BB * HH, nullptr);
    }

    // ---- layer 1
    k_zero_u32<<<256, 256, 0, stream>>>((unsigned int*)hA,   BB * HH / 2);
    k_zero_u32<<<256, 256, 0, stream>>>((unsigned int*)cbuf, BB * HH);
    k_xg_gemm<<<640, 128, 0, stream>>>(h1all, wih1h, bih1, bhh1, xgbuf, HH);
    for (int t = 0; t < S_LIVE; ++t) {
        const _Float16* hp = (t & 1) ? hB : hA;
        _Float16*       ho = (t & 1) ? hA : hB;
        k_lstm_step<<<128, 128, 0, stream>>>(xgbuf + (size_t)t * BB * G4, hp, whh1h,
                                             cbuf, ho, nullptr,
                                             (t == S_LIVE - 1) ? h2f32 : nullptr);
    }

    // ---- head on h2[4]
    k_head<<<16, 256, 0, stream>>>(h2f32, wout, bout, out);
}
